// MultiheadAttention_8650064134592
// MI455X (gfx1250) — compile-verified, matrix operand order checked by reading, not run on hardware
//
#include <hip/hip_runtime.h>
#include <hip/hip_bf16.h>

// ---------------------------------------------------------------------------
// MultiheadAttention forward for MI455X (gfx1250, wave32, WMMA).
// B=4, S=2048, E=1024, H=16, D=64.  d_out = [out (B*S*E f32)] ++ [attn (B*H*S*S f32)]
// ---------------------------------------------------------------------------

#define B_ 4
#define S_ 2048
#define E_ 1024
#define H_ 16
#define D_ 64
#define M_ (B_ * S_) /* 8192 rows for the projection GEMMs */

typedef __attribute__((ext_vector_type(16))) __bf16 v16bf;
typedef __attribute__((ext_vector_type(8)))  float  v8f;
typedef __attribute__((ext_vector_type(4)))  float  v4f;
typedef __attribute__((ext_vector_type(8)))  unsigned short v8u;
typedef __attribute__((ext_vector_type(16))) unsigned short v16u;
typedef __attribute__((ext_vector_type(2)))  unsigned v2u;

// f32 -> bf16, round-to-nearest-even
static __device__ __forceinline__ unsigned short f2bf(float f) {
  unsigned u = __builtin_bit_cast(unsigned, f);
  u += 0x7FFFu + ((u >> 16) & 1u);
  return (unsigned short)(u >> 16);
}
static __device__ __forceinline__ unsigned pack2(float a, float b) {
  return (unsigned)f2bf(a) | ((unsigned)f2bf(b) << 16);
}

// 16-byte async copy global -> LDS (ASYNCcnt-tracked, no VGPR round-trip).
// Generic LDS addresses carry the LDS byte offset in their low 32 bits.
static __device__ __forceinline__ void async_copy16(void* lds_dst, const void* gsrc) {
  unsigned lds = (unsigned)(uintptr_t)lds_dst;
  unsigned long long ga = (unsigned long long)(uintptr_t)gsrc;
  asm volatile("global_load_async_to_lds_b128 %0, %1, off"
               :: "v"(lds), "v"(ga) : "memory");
}
static __device__ __forceinline__ void async_wait0() {
  asm volatile("s_wait_asynccnt 0" ::: "memory");
}

// Load one 16x32 bf16 A/B WMMA fragment from a row-major LDS tile.
// Per the CDNA5 16-bit A-matrix layout, lane L (row = L&15) needs
// k = {kb..kb+7} U {kb+16..kb+23} with kb = (L>=16)?8:0 -> two 16B chunks.
static __device__ __forceinline__ v16bf ldfrag(const unsigned short* t, int stride, int k0) {
  int lane = threadIdx.x & 31;
  const unsigned short* p = t + (lane & 15) * stride + k0 + ((lane & 16) ? 8 : 0);
  v8u lo = *(const v8u*)(p);
  v8u hi = *(const v8u*)(p + 16);
  v16u r;
#pragma unroll
  for (int i = 0; i < 8; ++i) { r[i] = lo[i]; r[i + 8] = hi[i]; }
  return __builtin_bit_cast(v16bf, r);
}

static __device__ __forceinline__ v8f wmma_bf16(v16bf a, v16bf b, v8f c) {
  // (neg_a, A, neg_b, B, c_mod, C, reuse_a, reuse_b)
  return __builtin_amdgcn_wmma_f32_16x16x32_bf16(false, a, false, b, (short)0, c, false, false);
}

// ---------------------------------------------------------------------------
// GEMM: Y = X @ W^T + bias.  X:[M_ x 1024] f32, W:[1024 x 1024] f32 (rows = out cols).
// mode 0: bf16 head-split [B,H,S,D] (Q,K projections)
// mode 1: f32 into d_out (final output projection)
// mode 2: bf16 head-split TRANSPOSED [B,H,D,S] (V projection -> AV B-matrix layout)
// Block: 256 threads (8 waves), tile 128x64, BK=32. Each wave: 32x32 via 2x2 WMMA.
// ---------------------------------------------------------------------------
__global__ __launch_bounds__(256) void gemm_xwt_kernel(
    const float* __restrict__ X, const float* __restrict__ W,
    const float* __restrict__ bias, unsigned short* __restrict__ out_bf16_heads,
    float* __restrict__ out_f32, int mode) {
  __shared__ unsigned short Xs[128 * 40]; // 40-elem stride: 16B-aligned rows, spread banks
  __shared__ unsigned short Ws[64 * 40];
  const int K = E_, N = E_;
  int tid = threadIdx.x, lane = tid & 31, wave = tid >> 5;
  int M0 = blockIdx.x * 128, N0 = blockIdx.y * 64;
  int m0 = (wave >> 1) * 32, n0 = (wave & 1) * 32;
  v8f acc00 = {}, acc01 = {}, acc10 = {}, acc11 = {};

  for (int kt = 0; kt < K; kt += 32) {
    __syncthreads();
#pragma unroll
    for (int it = 0; it < 4; ++it) { // X tile: 128x32 f32 in 1024 float4 chunks
      int idx = tid + it * 256;
      int r = idx >> 3, c = (idx & 7) * 4;
      v4f x = *(const v4f*)&X[(size_t)(M0 + r) * K + kt + c];
      v2u p = {pack2(x[0], x[1]), pack2(x[2], x[3])};
      *(v2u*)&Xs[r * 40 + c] = p;
    }
#pragma unroll
    for (int it = 0; it < 2; ++it) { // W tile: 64x32 f32 in 512 float4 chunks
      int idx = tid + it * 256;
      int r = idx >> 3, c = (idx & 7) * 4;
      v4f x = *(const v4f*)&W[(size_t)(N0 + r) * K + kt + c];
      v2u p = {pack2(x[0], x[1]), pack2(x[2], x[3])};
      *(v2u*)&Ws[r * 40 + c] = p;
    }
    if (kt + 32 < K) // prefetch next X tile -> global_prefetch_b8
      __builtin_prefetch(&X[(size_t)(M0 + (tid >> 1)) * K + kt + 32 + (tid & 1) * 16], 0, 1);
    __syncthreads();

    v16bf a0 = ldfrag(Xs + (m0 +  0) * 40, 40, 0);
    v16bf a1 = ldfrag(Xs + (m0 + 16) * 40, 40, 0);
    v16bf b0 = ldfrag(Ws + (n0 +  0) * 40, 40, 0);
    v16bf b1 = ldfrag(Ws + (n0 + 16) * 40, 40, 0);
    acc00 = wmma_bf16(a0, b0, acc00);
    acc01 = wmma_bf16(a0, b1, acc01);
    acc10 = wmma_bf16(a1, b0, acc10);
    acc11 = wmma_bf16(a1, b1, acc11);
  }

  int mh = (lane & 16) ? 8 : 0;
  int nn = lane & 15;
#pragma unroll
  for (int i = 0; i < 2; ++i) {
#pragma unroll
    for (int j = 0; j < 2; ++j) {
      v8f acc = (i == 0) ? (j == 0 ? acc00 : acc01) : (j == 0 ? acc10 : acc11);
      int n = N0 + n0 + j * 16 + nn;       // fixed per lane
      int mbase = M0 + m0 + i * 16 + mh;   // rows mbase..mbase+7 (same batch b)
      if (mode == 0) {
        int h = n >> 6, d = n & 63;
#pragma unroll
        for (int r = 0; r < 8; ++r) {
          int m = mbase + r, b = m >> 11, s = m & (S_ - 1);
          out_bf16_heads[(((size_t)(b * H_ + h) * S_) + s) * D_ + d] = f2bf(acc[r] + bias[n]);
        }
      } else if (mode == 1) {
#pragma unroll
        for (int r = 0; r < 8; ++r)
          out_f32[(size_t)(mbase + r) * N + n] = acc[r] + bias[n];
      } else { // mode 2: transposed head-split, one packed 16B store per lane
        int h = n >> 6, d = n & 63;
        int b = mbase >> 11, s0 = mbase & (S_ - 1);
        v8u pk;
#pragma unroll
        for (int r = 0; r < 8; ++r) pk[r] = f2bf(acc[r] + bias[n]);
        *(v8u*)&out_bf16_heads[(((size_t)(b * H_ + h) * D_) + d) * S_ + s0] = pk;
      }
    }
  }
}

// ---------------------------------------------------------------------------
// Attention: per block = one (b,h) and one 16-row q tile.
// Phase 1: logits (Q@K^T * 1/8) for all 2048 keys into LDS (f32).
// Phase 2: row softmax in LDS; write attn f32 to global + bf16 copy in LDS.
// Phase 3: vals = attn @ V via WMMA from LDS; f32 vals to workspace.
// K / V^T staged with GLOBAL_LOAD_ASYNC_TO_LDS_B128 (ASYNCcnt).
// Dynamic LDS ~218 KB (320 KB/WGP available on CDNA5).
// ---------------------------------------------------------------------------
#define LSTR 2056 /* f32 logits row stride    */
#define QSTR 72   /* q-tile bf16 row stride   */
#define KSTR 72   /* k-stage bf16 row stride  */
#define VSTR 136  /* v^T-stage bf16 row stride */
#define SKB  128  /* keys staged per sweep step */

__global__ __launch_bounds__(256) void attn_kernel(
    const unsigned short* __restrict__ Qh, const unsigned short* __restrict__ Kh,
    const unsigned short* __restrict__ Vt, float* __restrict__ attn_out,
    float* __restrict__ vals_out) {
  extern __shared__ char smem[];
  float*          Llog = (float*)smem;                                    // 16*LSTR f32
  unsigned short* Latt = (unsigned short*)(smem + 16 * LSTR * 4);         // 16*LSTR bf16
  unsigned short* Lq   = (unsigned short*)(smem + 16 * LSTR * 4 + 16 * LSTR * 2);
  unsigned short* Lkv  = Lq + 16 * QSTR;                                  // K stage / V^T stage
  float*          Lred = Llog;                                            // reuse (dead after ph.2)

  int tid = threadIdx.x, lane = tid & 31, wave = tid >> 5;
  int bh = blockIdx.y, b = bh >> 4, h = bh & 15;
  int q0 = blockIdx.x * 16;

  const unsigned short* Qb  = Qh + ((size_t)bh * S_ + q0) * D_;
  const unsigned short* Kb  = Kh + (size_t)bh * S_ * D_;
  const unsigned short* Vtb = Vt + (size_t)bh * D_ * S_;   // [d][s]

  // stage Q (16x64 bf16 = 128 x 16B chunks) asynchronously
  if (tid < 128) {
    int r = tid >> 3, c = (tid & 7) * 8;
    async_copy16(&Lq[r * QSTR + c], Qb + r * D_ + c);
  }
  async_wait0();
  __syncthreads();
  v16bf qa0 = ldfrag(Lq, QSTR, 0);
  v16bf qa1 = ldfrag(Lq, QSTR, 32);

  // ---- Phase 1: logits ----
  for (int kb = 0; kb < S_ / SKB; ++kb) {
    __syncthreads();
#pragma unroll
    for (int it = 0; it < 4; ++it) { // K block: 128x64 bf16 = 1024 x 16B chunks
      int idx = tid + it * 256;
      int r = idx >> 3, c = (idx & 7) * 8;
      async_copy16(&Lkv[r * KSTR + c], Kb + (size_t)(kb * SKB + r) * D_ + c);
    }
    async_wait0();
    __syncthreads();
    int n0 = wave * 16; // each wave owns one 16-key tile of the 128
    v16bf b0 = ldfrag(Lkv + n0 * KSTR, KSTR, 0);
    v16bf b1 = ldfrag(Lkv + n0 * KSTR, KSTR, 32);
    v8f lacc = {};
    lacc = wmma_bf16(qa0, b0, lacc);
    lacc = wmma_bf16(qa1, b1, lacc);
    int mh = (lane & 16) ? 8 : 0;
#pragma unroll
    for (int r = 0; r < 8; ++r)
      Llog[(r + mh) * LSTR + kb * SKB + n0 + (lane & 15)] = lacc[r] * 0.125f; // 1/sqrt(64)
  }
  __syncthreads();

  // ---- Phase 2: softmax (each wave: rows 2w, 2w+1) ----
#pragma unroll
  for (int rr = 0; rr < 2; ++rr) {
    int row = wave * 2 + rr;
    float* Lr = Llog + row * LSTR;
    float mx = -1e30f;
    for (int c = lane; c < S_; c += 32) mx = fmaxf(mx, Lr[c]);
#pragma unroll
    for (int off = 16; off; off >>= 1) mx = fmaxf(mx, __shfl_xor(mx, off, 32));
    float sum = 0.f;
    for (int c = lane; c < S_; c += 32) { float e = __expf(Lr[c] - mx); Lr[c] = e; sum += e; }
#pragma unroll
    for (int off = 16; off; off >>= 1) sum += __shfl_xor(sum, off, 32);
    float inv = 1.f / sum;
    float* grow = attn_out + ((size_t)bh * S_ + (q0 + row)) * S_;
    for (int c = lane; c < S_; c += 32) {
      float a = Lr[c] * inv;
      grow[c] = a;                     // write attn once (write-only stream)
      Latt[row * LSTR + c] = f2bf(a);  // bf16 copy for the AV WMMAs
    }
  }

  // ---- Phase 3: vals = attn @ V ----
  int tileN = wave & 3;  // which 16-wide d slice (D=64 -> 4 tiles)
  int half  = wave >> 2; // wave pairs split the 128 staged keys
  v8f vac = {};
  for (int kb = 0; kb < S_ / SKB; ++kb) {
    __syncthreads();
#pragma unroll
    for (int it = 0; it < 4; ++it) { // V^T block: 64 d-rows x 128 keys = 1024 x 16B chunks
      int idx = tid + it * 256;
      int r = idx >> 4, c = (idx & 15) * 8; // r = d row, c = key offset
      async_copy16(&Lkv[r * VSTR + c], Vtb + (size_t)r * S_ + kb * SKB + c);
    }
    async_wait0();
    __syncthreads();
#pragma unroll
    for (int ks = 0; ks < 2; ++ks) {
      int kl = half * 64 + ks * 32;
      v16bf a  = ldfrag(Latt, LSTR, kb * SKB + kl);
      v16bf bm = ldfrag(Lkv + tileN * 16 * VSTR, VSTR, kl);
      vac = wmma_bf16(a, bm, vac);
    }
  }
  __syncthreads();
  int mh = (lane & 16) ? 8 : 0, nn = lane & 15;
  if (half == 1) { // partials from upper key half
#pragma unroll
    for (int r = 0; r < 8; ++r)
      Lred[tileN * 256 + (r + mh) * 16 + nn] = vac[r];
  }
  __syncthreads();
  if (half == 0) {
#pragma unroll
    for (int r = 0; r < 8; ++r) {
      float v = vac[r] + Lred[tileN * 256 + (r + mh) * 16 + nn];
      int m = q0 + r + mh;                 // sequence index s
      int n = h * D_ + tileN * 16 + nn;    // merged-head column
      vals_out[((size_t)b * S_ + m) * E_ + n] = v;
    }
  }
}

// ---------------------------------------------------------------------------
extern "C" void kernel_launch(void* const* d_in, const int* in_sizes, int n_in,
                              void* d_out, int out_size, void* d_ws, size_t ws_size,
                              hipStream_t stream) {
  (void)in_sizes; (void)n_in; (void)out_size; (void)ws_size;
  const float* q  = (const float*)d_in[0];
  const float* k  = (const float*)d_in[1];
  const float* v  = (const float*)d_in[2];
  const float* Wq = (const float*)d_in[3];
  const float* Wk = (const float*)d_in[4];
  const float* Wv = (const float*)d_in[5];
  const float* Wo = (const float*)d_in[6];
  const float* bq = (const float*)d_in[7];
  const float* bk = (const float*)d_in[8];
  const float* bv = (const float*)d_in[9];
  const float* bo = (const float*)d_in[10];

  float* out  = (float*)d_out;
  float* attn = out + (size_t)B_ * S_ * E_; // tuple: (out, attn) concatenated

  // Workspace: Qh/Kh bf16 [B,H,S,D], Vt bf16 [B,H,D,S] (16MB each) + vals f32 (32MB).
  size_t nElem = (size_t)B_ * S_ * E_;
  unsigned short* Qh = (unsigned short*)d_ws;
  unsigned short* Kh = Qh + nElem;
  unsigned short* Vt = Kh + nElem;
  float* vals = (float*)(Vt + nElem);

  dim3 ggrid(M_ / 128, E_ / 64);
  gemm_xwt_kernel<<<ggrid, 256, 0, stream>>>(q, Wq, bq, Qh, nullptr, 0);
  gemm_xwt_kernel<<<ggrid, 256, 0, stream>>>(k, Wk, bk, Kh, nullptr, 0);
  gemm_xwt_kernel<<<ggrid, 256, 0, stream>>>(v, Wv, bv, Vt, nullptr, 2);

  int ldsBytes = 16 * LSTR * 4 + 16 * LSTR * 2 + 16 * QSTR * 2 + SKB * KSTR * 2; // ~218KB
  (void)hipFuncSetAttribute(reinterpret_cast<const void*>(attn_kernel),
                            hipFuncAttributeMaxDynamicSharedMemorySize, ldsBytes);
  attn_kernel<<<dim3(S_ / 16, B_ * H_), 256, ldsBytes, stream>>>(Qh, Kh, Vt, attn, vals);

  gemm_xwt_kernel<<<ggrid, 256, 0, stream>>>(vals, Wo, bo, nullptr, out, 1);
}
